// S_Layer_26938034881042
// MI455X (gfx1250) — matrix operand order repaired and checked by reading, not compiled, not run on hardware
//
#include <hip/hip_runtime.h>
#include <math.h>

typedef __attribute__((ext_vector_type(16))) _Float16 v16h;
typedef __attribute__((ext_vector_type(8)))  _Float16 v8h;
typedef __attribute__((ext_vector_type(8)))  float    v8f;
typedef __attribute__((ext_vector_type(4)))  float    v4f;

#define N_ 128
#define D_ 512
#define L_ 512
#define K_ 512

static __device__ __forceinline__ float denomf() { return (float)(L_ - 1); }

// ---------------------------------------------------------------------------
// Kernel 1: precompute basis (f16, transposed, l-contiguous) + window scale.
//   bcT[kout][l] = cos(2*pi/(L-1) * l * kout)
//   bsT[kout][l] = sin(2*pi/(L-1) * l * kout)
//   wsc[d][l]    = (a[d] - (1-a[d])*cos(2*pi*l/(L-1))) * exp(sigma[d]/(L-1))
// ---------------------------------------------------------------------------
__global__ void precompute_kernel(const float* __restrict__ a,
                                  const float* __restrict__ sigma,
                                  _Float16* __restrict__ bcT,
                                  _Float16* __restrict__ bsT,
                                  float* __restrict__ wsc) {
  const int idx = blockIdx.x * blockDim.x + threadIdx.x;   // 0 .. 512*512-1
  const int l = idx & (L_ - 1);
  const int r = idx >> 9;                                  // kout / d
  const float w0 = 6.283185307179586f / denomf();
  const float ang = w0 * (float)l * (float)r;
  bcT[idx] = (_Float16)cosf(ang);
  bsT[idx] = (_Float16)sinf(ang);
  const float ad = a[r];
  const float win = ad - (1.0f - ad) * cosf(w0 * (float)l);
  wsc[idx] = win * expf(sigma[r] / denomf());
}

// ---------------------------------------------------------------------------
// Kernel 2 (fast path only): xh[row][l] = (f16)(x[row][l] * wsc[row%D][l])
// 8 elements / thread, pure streaming pass (HBM-bandwidth bound).
// ---------------------------------------------------------------------------
__global__ __launch_bounds__(256)
void scale_kernel(const float* __restrict__ x,
                  const float* __restrict__ wsc,
                  _Float16* __restrict__ xh) {
  const size_t base = ((size_t)blockIdx.x * blockDim.x + threadIdx.x) * 8u;
  const unsigned l = (unsigned)(base & (L_ - 1u));
  const unsigned d = (unsigned)((base >> 9) & (D_ - 1u));
  const v4f x0 = *(const v4f*)(x + base);
  const v4f x1 = *(const v4f*)(x + base + 4);
  const v4f w0 = *(const v4f*)(wsc + (size_t)d * L_ + l);
  const v4f w1 = *(const v4f*)(wsc + (size_t)d * L_ + l + 4);
  v8h h;
#pragma unroll
  for (int j = 0; j < 4; ++j) {
    h[j]     = (_Float16)(x0[j] * w0[j]);
    h[j + 4] = (_Float16)(x1[j] * w1[j]);
  }
  *(v8h*)(xh + base) = h;
}

// ---------------------------------------------------------------------------
// Fragment bundle for one K-step of 32 (A + 4 B fragments).
// ---------------------------------------------------------------------------
struct Frags {
  v16h af, bc0, bs0, bc1, bs1;
};

static __device__ __forceinline__ Frags
load_frags(const _Float16* __restrict__ xhrow, unsigned kbA,
           const _Float16* __restrict__ bcp,
           const _Float16* __restrict__ bsp, unsigned l0) {
  Frags f;
  const v8h a0 = *(const v8h*)(xhrow + l0 + kbA);
  const v8h a1 = *(const v8h*)(xhrow + l0 + kbA + 8);
  const v8h c0a = *(const v8h*)(bcp + l0);
  const v8h c0b = *(const v8h*)(bcp + l0 + 8);
  const v8h s0a = *(const v8h*)(bsp + l0);
  const v8h s0b = *(const v8h*)(bsp + l0 + 8);
  const v8h c1a = *(const v8h*)(bcp + (size_t)16 * L_ + l0);
  const v8h c1b = *(const v8h*)(bcp + (size_t)16 * L_ + l0 + 8);
  const v8h s1a = *(const v8h*)(bsp + (size_t)16 * L_ + l0);
  const v8h s1b = *(const v8h*)(bsp + (size_t)16 * L_ + l0 + 8);
#pragma unroll
  for (int j = 0; j < 8; ++j) {
    f.af[j]  = a0[j];  f.af[j + 8]  = a1[j];
    f.bc0[j] = c0a[j]; f.bc0[j + 8] = c0b[j];
    f.bs0[j] = s0a[j]; f.bs0[j + 8] = s0b[j];
    f.bc1[j] = c1a[j]; f.bc1[j + 8] = c1b[j];
    f.bs1[j] = s1a[j]; f.bs1[j + 8] = s1b[j];
  }
  return f;
}

// ---------------------------------------------------------------------------
// Kernel 3: dual-basis WMMA GEMM + magnitude epilogue.
// One wave -> one 16x32 output slab (two 16x16 col tiles), 4 accumulators
// (cos/sin x 2 tiles) reusing one A fragment -> 4 WMMAs per K-step of 32.
// PRESCALED path is software-pipelined: fragments for step i+1 are issued
// before the WMMAs of step i, so WMMA latency hides the load round-trip.
// ---------------------------------------------------------------------------
template <bool PRESCALED>
__global__ __launch_bounds__(256)
void dft_mag_gemm(const float* __restrict__ x,
                  const _Float16* __restrict__ xh,
                  const float* __restrict__ wsc,
                  const _Float16* __restrict__ bcT,
                  const _Float16* __restrict__ bsT,
                  float* __restrict__ out) {
  const unsigned lane    = threadIdx.x & 31u;
  const unsigned wave    = threadIdx.x >> 5;
  const unsigned slab    = blockIdx.x * 8u + wave;    // 16x32 output slabs
  const unsigned colPair = slab & 15u;                // 16 pairs of col tiles
  const unsigned rowTile = slab >> 4;                 // 4096 row tiles

  const bool     hiHalf = (lane >= 16u);
  const unsigned m      = lane & 15u;                 // A row within tile
  const unsigned row    = rowTile * 16u + m;          // global row in [0, N*D)
  const unsigned d      = row & (D_ - 1u);
  const unsigned nc     = lane & 15u;                 // B/D column within tile
  const unsigned kout0  = colPair * 32u + nc;         // first col tile column
  const unsigned kbA    = hiHalf ? 16u : 0u;          // A K sub-offset
  const unsigned koffB  = hiHalf ? 16u : 0u;          // B K sub-offset

  const float*    xrow  = x   + (size_t)row * L_;
  const _Float16* xhrow = xh  + (size_t)row * L_;
  const float*    wrow  = wsc + (size_t)d   * L_;
  const _Float16* bcp   = bcT + (size_t)kout0 * L_ + koffB;  // tile0; tile1 = +16*L_
  const _Float16* bsp   = bsT + (size_t)kout0 * L_ + koffB;

  v8f accC0 = {}, accS0 = {}, accC1 = {}, accS1 = {};

  if constexpr (PRESCALED) {
    // ---- double-buffered pipeline over 16 K-steps ----
    Frags cur = load_frags(xhrow, kbA, bcp, bsp, 0u);
#pragma unroll
    for (unsigned l0 = 0; l0 < L_ - 32u; l0 += 32u) {
      Frags nxt = load_frags(xhrow, kbA, bcp, bsp, l0 + 32u);
      accC0 = __builtin_amdgcn_wmma_f32_16x16x32_f16(
          false, cur.af, false, cur.bc0, (short)0, accC0, false, false);
      accS0 = __builtin_amdgcn_wmma_f32_16x16x32_f16(
          false, cur.af, false, cur.bs0, (short)0, accS0, false, false);
      accC1 = __builtin_amdgcn_wmma_f32_16x16x32_f16(
          false, cur.af, false, cur.bc1, (short)0, accC1, false, false);
      accS1 = __builtin_amdgcn_wmma_f32_16x16x32_f16(
          false, cur.af, false, cur.bs1, (short)0, accS1, false, false);
      cur = nxt;
    }
    accC0 = __builtin_amdgcn_wmma_f32_16x16x32_f16(
        false, cur.af, false, cur.bc0, (short)0, accC0, false, false);
    accS0 = __builtin_amdgcn_wmma_f32_16x16x32_f16(
        false, cur.af, false, cur.bs0, (short)0, accS0, false, false);
    accC1 = __builtin_amdgcn_wmma_f32_16x16x32_f16(
        false, cur.af, false, cur.bc1, (short)0, accC1, false, false);
    accS1 = __builtin_amdgcn_wmma_f32_16x16x32_f16(
        false, cur.af, false, cur.bs1, (short)0, accS1, false, false);
  } else {
    // ---- fallback: build A fragment on the fly from fp32 x * wscale ----
    for (unsigned l0 = 0; l0 < L_; l0 += 32u) {
      const unsigned cA = l0 + kbA;
      const unsigned cB = cA + 8u;
      const v4f xa0 = *(const v4f*)(xrow + cA);
      const v4f xa1 = *(const v4f*)(xrow + cA + 4);
      const v4f xb0 = *(const v4f*)(xrow + cB);
      const v4f xb1 = *(const v4f*)(xrow + cB + 4);
      const v4f wa0 = *(const v4f*)(wrow + cA);
      const v4f wa1 = *(const v4f*)(wrow + cA + 4);
      const v4f wb0 = *(const v4f*)(wrow + cB);
      const v4f wb1 = *(const v4f*)(wrow + cB + 4);
      v16h af;
#pragma unroll
      for (int j = 0; j < 4; ++j) {
        af[j]      = (_Float16)(xa0[j] * wa0[j]);
        af[j + 4]  = (_Float16)(xa1[j] * wa1[j]);
        af[j + 8]  = (_Float16)(xb0[j] * wb0[j]);
        af[j + 12] = (_Float16)(xb1[j] * wb1[j]);
      }
      const v8h c0a = *(const v8h*)(bcp + l0);
      const v8h c0b = *(const v8h*)(bcp + l0 + 8);
      const v8h s0a = *(const v8h*)(bsp + l0);
      const v8h s0b = *(const v8h*)(bsp + l0 + 8);
      const v8h c1a = *(const v8h*)(bcp + (size_t)16 * L_ + l0);
      const v8h c1b = *(const v8h*)(bcp + (size_t)16 * L_ + l0 + 8);
      const v8h s1a = *(const v8h*)(bsp + (size_t)16 * L_ + l0);
      const v8h s1b = *(const v8h*)(bsp + (size_t)16 * L_ + l0 + 8);
      v16h bc0, bs0, bc1, bs1;
#pragma unroll
      for (int j = 0; j < 8; ++j) {
        bc0[j] = c0a[j]; bc0[j + 8] = c0b[j];
        bs0[j] = s0a[j]; bs0[j + 8] = s0b[j];
        bc1[j] = c1a[j]; bc1[j + 8] = c1b[j];
        bs1[j] = s1a[j]; bs1[j + 8] = s1b[j];
      }
      accC0 = __builtin_amdgcn_wmma_f32_16x16x32_f16(
          false, af, false, bc0, (short)0, accC0, false, false);
      accS0 = __builtin_amdgcn_wmma_f32_16x16x32_f16(
          false, af, false, bs0, (short)0, accS0, false, false);
      accC1 = __builtin_amdgcn_wmma_f32_16x16x32_f16(
          false, af, false, bc1, (short)0, accC1, false, false);
      accS1 = __builtin_amdgcn_wmma_f32_16x16x32_f16(
          false, af, false, bs1, (short)0, accS1, false, false);
    }
  }

  // ---- epilogue: magnitude; D layout: VGPR i -> row i (lanes 0-15) / i+8 ----
  const unsigned mBase = hiHalf ? 8u : 0u;
#pragma unroll
  for (int i = 0; i < 8; ++i) {
    const unsigned r = rowTile * 16u + mBase + (unsigned)i;
    float cr = accC0[i], si = accS0[i];
    out[(size_t)r * K_ + kout0] = sqrtf(cr * cr + si * si);
    cr = accC1[i]; si = accS1[i];
    out[(size_t)r * K_ + kout0 + 16] = sqrtf(cr * cr + si * si);
  }
}

// ---------------------------------------------------------------------------
extern "C" void kernel_launch(void* const* d_in, const int* in_sizes, int n_in,
                              void* d_out, int out_size, void* d_ws, size_t ws_size,
                              hipStream_t stream) {
  const float* x     = (const float*)d_in[0];
  const float* a     = (const float*)d_in[1];
  const float* sigma = (const float*)d_in[2];
  float* out = (float*)d_out;

  // workspace layout: bcT (512KB f16) | bsT (512KB f16) | wsc (1MB f32) | xh (64MB f16)
  _Float16* bcT = (_Float16*)d_ws;
  _Float16* bsT = bcT + (size_t)L_ * K_;
  float*    wsc = (float*)(bsT + (size_t)L_ * K_);
  _Float16* xh  = (_Float16*)(wsc + (size_t)D_ * L_);

  const size_t needSlow = (size_t)L_ * K_ * 2 * sizeof(_Float16)
                        + (size_t)D_ * L_ * sizeof(float);
  const size_t needFast = needSlow + (size_t)N_ * D_ * L_ * sizeof(_Float16);

  precompute_kernel<<<(512 * L_) / 256, 256, 0, stream>>>(a, sigma, bcT, bsT, wsc);

  const unsigned slabs = ((N_ * D_) / 16) * (K_ / 32);   // 4096 * 16 = 65536
  if (ws_size >= needFast) {
    scale_kernel<<<(N_ * D_ * L_) / (8 * 256), 256, 0, stream>>>(x, wsc, xh);
    dft_mag_gemm<true><<<slabs / 8, 256, 0, stream>>>(x, xh, wsc, bcT, bsT, out);
  } else {
    dft_mag_gemm<false><<<slabs / 8, 256, 0, stream>>>(x, xh, wsc, bcT, bsT, out);
  }
}
